// ResNet18_quant_20572893348557
// MI455X (gfx1250) — matrix operand order repaired and checked by reading, not compiled, not run on hardware
//
#include <hip/hip_runtime.h>
#include <hip/hip_bf16.h>
#include <stdint.h>

// ---------------------------------------------------------------------------
// ResNet18-quant inference for MI455X (gfx1250, wave32, WMMA).
// Activations kept as f16 NHWC. Convs run as implicit GEMM with
// v_wmma_f32_16x16x32_f16, LDS double buffering, fused BN/residual/ReLU.
// B (weight) tiles copied to LDS with GLOBAL_LOAD_ASYNC_TO_LDS_B128 when the
// toolchain exposes the builtin (ASYNCcnt path), else synchronous fallback.
// All spatial decompositions use shifts (every H*W here is a power of two).
// ---------------------------------------------------------------------------

typedef __attribute__((ext_vector_type(16))) _Float16 v16h;
typedef __attribute__((ext_vector_type(8)))  float    v8f;

// 128-bit payload type for the async-LDS builtin (param type per hipcc
// diagnostic: pointer to int __vector(4), address-space qualified).
typedef int b128_t __attribute__((vector_size(16)));
typedef __attribute__((address_space(1))) b128_t* global_b128_ptr;
typedef __attribute__((address_space(3))) b128_t* lds_b128_ptr;

#define TM 128          // block tile M (output positions)
#define TN 64           // block tile N (output channels)
#define TK 32           // K step (matches 16x16x32 WMMA)
#define TKA_PAD 34      // A row pad: 68B stride (17 banks, coprime with 64)

#if __has_builtin(__builtin_amdgcn_global_load_async_to_lds_b128) && \
    __has_builtin(__builtin_amdgcn_s_wait_asynccnt)
#define HAVE_ASYNC_LDS 1
#else
#define HAVE_ASYNC_LDS 0
#endif

// ------------------------------ prep kernels -------------------------------

// per-tensor absmax -> scale = absmax/127 (single block, grid-stride)
__global__ __launch_bounds__(1024) void absmax_scale_kernel(
    const float* __restrict__ w, int n, float* __restrict__ scale_out) {
  __shared__ float red[1024];
  float m = 0.f;
  for (int i = threadIdx.x; i < n; i += blockDim.x) m = fmaxf(m, fabsf(w[i]));
  red[threadIdx.x] = m;
  __syncthreads();
  for (int s = 512; s > 0; s >>= 1) {
    if (threadIdx.x < s) red[threadIdx.x] = fmaxf(red[threadIdx.x], red[threadIdx.x + s]);
    __syncthreads();
  }
  if (threadIdx.x == 0) scale_out[0] = red[0] / 127.0f;
}

// OIHW f32 -> [K = (kh*KW+kw)*Cin+ci][Cout] f16, optional 8-bit fake quant
__global__ void quant_pack_kernel(const float* __restrict__ w,
                                  const float* __restrict__ scale_p,
                                  _Float16* __restrict__ B,
                                  int Cout, int Cin, int KH, int KW, int quant) {
  int idx = blockIdx.x * blockDim.x + threadIdx.x;
  int total = Cout * Cin * KH * KW;
  if (idx >= total) return;
  int kw = idx % KW; int t = idx / KW;
  int kh = t % KH;   t /= KH;
  int ci = t % Cin;  int co = t / Cin;
  float v = w[idx];
  if (quant) {
    float s = scale_p[0];
    if (s > 0.f) {
      float q = rintf(v / s);
      q = fminf(fmaxf(q, -128.f), 127.f);
      v = q * s;
    }
  }
  int k = (kh * KW + kw) * Cin + ci;
  B[(size_t)k * Cout + co] = (_Float16)v;
}

// fold conv bias + BN into per-channel scale/bias:  y = acc*scale + bias
__global__ void bn_fold_kernel(const float* __restrict__ b, const float* __restrict__ g,
                               const float* __restrict__ be, const float* __restrict__ mu,
                               const float* __restrict__ var,
                               float* __restrict__ scale, float* __restrict__ bias, int C) {
  int c = blockIdx.x * blockDim.x + threadIdx.x;
  if (c >= C) return;
  float inv = g[c] * rsqrtf(var[c] + 1e-5f);
  scale[c] = inv;
  bias[c]  = (b[c] - mu[c]) * inv + be[c];
}

// NCHW f32 -> NHWC f16 (N=256, C=3, H=W=32 fixed sizes; pow2 spatial)
__global__ void nchw_to_nhwc_h_kernel(const float* __restrict__ x, _Float16* __restrict__ y,
                                      int N, int C, int H, int W) {
  int idx = blockIdx.x * blockDim.x + threadIdx.x;
  int total = N * C * H * W;
  if (idx >= total) return;
  int w = idx & (W - 1); int t = idx >> 5;         // W = 32
  int h = t & (H - 1);   t >>= 5;                  // H = 32
  int c = t % C;   int n = t / C;
  y[(((size_t)n * H + h) * W + w) * C + c] = (_Float16)x[idx];
}

// 3x3 stride-2 pad-1 maxpool on NHWC f16 (C, W, Ho, Wo all pow2)
__global__ void maxpool3x3s2_kernel(const _Float16* __restrict__ in, _Float16* __restrict__ out,
                                    int N, int H, int W, int C, int Ho, int Wo) {
  int idx = blockIdx.x * blockDim.x + threadIdx.x;
  int total = N * Ho * Wo * C;
  if (idx >= total) return;
  int c = idx & (C - 1); int t = idx / C;
  int wo = t & (Wo - 1); t /= Wo;
  int ho = t & (Ho - 1); int n = t / Ho;
  float m = -3.0e38f;
  #pragma unroll
  for (int dy = 0; dy < 3; ++dy) {
    int hi = ho * 2 - 1 + dy;
    if ((unsigned)hi >= (unsigned)H) continue;
    #pragma unroll
    for (int dx = 0; dx < 3; ++dx) {
      int wi = wo * 2 - 1 + dx;
      if ((unsigned)wi >= (unsigned)W) continue;
      m = fmaxf(m, (float)in[(((size_t)n * H + hi) * W + wi) * C + c]);
    }
  }
  out[idx] = (_Float16)m;
}

// adaptive avgpool (HW positions) + FC (C -> ncls), one block per image
__global__ __launch_bounds__(128) void avgpool_fc_kernel(
    const _Float16* __restrict__ in, const float* __restrict__ fw,
    const float* __restrict__ fb, float* __restrict__ out,
    int HW, int C, int ncls) {
  __shared__ float pool[512];
  int n = blockIdx.x;
  for (int c = threadIdx.x; c < C; c += blockDim.x) {
    float s = 0.f;
    for (int p = 0; p < HW; ++p) s += (float)in[((size_t)n * HW + p) * C + c];
    pool[c] = s / (float)HW;
  }
  __syncthreads();
  for (int o = threadIdx.x; o < ncls; o += blockDim.x) {
    float s = fb[o];
    for (int c = 0; c < C; ++c) s += pool[c] * fw[c * ncls + o];
    out[(size_t)n * ncls + o] = s;
  }
}

// ------------------------- implicit-GEMM conv (WMMA) -----------------------
//
// out[m, co] = relu( (sum_k A[m,k] * B[k,co]) * scale[co] + bias[co] + resid )
// A: im2col patches from NHWC f16 input, k = (kh*KW+kw)*Cin + ci
// B: packed weights [K][Cout] f16; LDS B tile kept in natural [k][n] layout
//    (row = 64 couts = 128B, so per-lane 16B chunks are 16B-aligned ->
//     eligible for GLOBAL_LOAD_ASYNC_TO_LDS_B128)
// Block: 256 threads = 8 waves (wave32); block tile 128x64, per-wave 32x32
// (2x2 accumulators of 16x16), K stepped by 32 via v_wmma_f32_16x16x32_f16.
// Fragment layouts per CDNA5 ISA 7.12.2:
//   A 16x32 f16 : lane l (m = l&15) holds K {0..7,16..23} (+8 for l>=16)
//   B 32x16 f16 : lane l (n = l&15) holds K {0..7,16..23} (+8 for l>=16)
//   C/D 16x16 f32: VGPR r -> m = r + 8*(l>>4), n = l&15

union FragA { v16h v; _Float16 h[16]; };
union FragC { v8f  v; float    f[8]; };

template<int KHW, bool POW2>
__global__ __launch_bounds__(256) void conv_wmma_kernel(
    const _Float16* __restrict__ in, const _Float16* __restrict__ wgt,
    const float* __restrict__ scale, const float* __restrict__ bias,
    const _Float16* __restrict__ resid, _Float16* __restrict__ out,
    int H, int W, int Cin, int Cout, int stride, int pad,
    int relu, int cinShift, int hwShift, int woShift) {
  __shared__ _Float16 As[2][TM][TKA_PAD];
  __shared__ _Float16 Bs[2][TK][TN];       // natural [k][n]; row = 128B

  const int tid  = threadIdx.x;
  const int lane = tid & 31;
  const int wv   = tid >> 5;
  const int wm   = wv & 3;     // M offset 32*wm within block tile
  const int wn   = wv >> 2;    // N offset 32*wn within block tile
  const int m0   = blockIdx.x * TM;
  const int n0   = blockIdx.y * TN;
  const int K    = Cin * KHW * KHW;

  // ---- A loader: 2 threads per row, 16 contiguous k each ----
  // All Ho*Wo and Wo here are powers of two -> shifts instead of divisions.
  const int arow = tid >> 1;
  const int aseg = (tid & 1) * 16;
  const int am   = m0 + arow;               // M is always a multiple of 128 here
  const int an   = am >> hwShift;
  const int arem = am & ((1 << hwShift) - 1);
  const int aho  = arem >> woShift;
  const int awo  = arem & ((1 << woShift) - 1);
  const int ahb  = aho * stride - pad;
  const int awb  = awo * stride - pad;

  // ---- B loader: 8 threads per k-row, 8 contiguous couts (16B) each ----
  const int bk  = tid >> 3;                 // 0..31
  const int bco = (tid & 7) * 8;            // 0..56

  FragC acc[2][2];
  #pragma unroll
  for (int mi = 0; mi < 2; ++mi)
    #pragma unroll
    for (int ni = 0; ni < 2; ++ni)
      #pragma unroll
      for (int r = 0; r < 8; ++r) acc[mi][ni].f[r] = 0.f;

  auto load_stage = [&](int k0, int buf) {
    // ---- A tile (im2col on the fly) ----
    if (POW2) {
      // Cin is a power of two >= 64: a 16-run of k shares (kh,kw) and is
      // channel-contiguous in NHWC -> vectorizable global loads.
      const int kbase = k0 + aseg;          // POW2 => K % 32 == 0, kbase < K
      const int ci0 = kbase & (Cin - 1);
      const int r   = kbase >> cinShift;
      const int kw  = r % KHW;
      const int kh  = r / KHW;
      const int hi  = ahb + kh, wi = awb + kw;
      if ((unsigned)hi < (unsigned)H && (unsigned)wi < (unsigned)W) {
        const _Float16* src = in + (((size_t)an * H + hi) * W + wi) * Cin + ci0;
        #pragma unroll
        for (int j = 0; j < 16; ++j) As[buf][arow][aseg + j] = src[j];
      } else {
        #pragma unroll
        for (int j = 0; j < 16; ++j) As[buf][arow][aseg + j] = (_Float16)0.f;
      }
    } else {
      // General path: only the stem uses this (Cin == 3); literal 3 lets the
      // compiler use magic-number sequences instead of runtime division.
      #pragma unroll
      for (int j = 0; j < 16; ++j) {
        const int k = k0 + aseg + j;
        _Float16 v = (_Float16)0.f;
        if (k < K) {
          const int ci = k % 3;
          const int r  = k / 3;
          const int kw = r % KHW;
          const int kh = r / KHW;
          const int hi = ahb + kh, wi = awb + kw;
          if ((unsigned)hi < (unsigned)H && (unsigned)wi < (unsigned)W)
            v = in[(((size_t)an * H + hi) * W + wi) * Cin + ci];
        }
        As[buf][arow][aseg + j] = v;
      }
    }
    // ---- B tile: [k][cout] global -> [k][n] LDS, 16B per lane ----
#if HAVE_ASYNC_LDS
    if (POW2) {
      // K % 32 == 0 in every POW2 conv -> no partial k rows; one async
      // b128 per lane (512B per wave, 4KB per stage) tracked by ASYNCcnt.
      const _Float16* gsrc = wgt + (size_t)(k0 + bk) * Cout + n0 + bco;
      __builtin_amdgcn_global_load_async_to_lds_b128(
          (global_b128_ptr)(void*)gsrc,
          (lds_b128_ptr)(void*)&Bs[buf][bk][bco],
          0, 0);
      return;
    }
#endif
    const int k = k0 + bk;
    if (k < K) {
      const _Float16* src = wgt + (size_t)k * Cout + n0 + bco;
      #pragma unroll
      for (int j = 0; j < 8; ++j) Bs[buf][bk][bco + j] = src[j];
    } else {
      #pragma unroll
      for (int j = 0; j < 8; ++j) Bs[buf][bk][bco + j] = (_Float16)0.f;
    }
  };

  load_stage(0, 0);
#if HAVE_ASYNC_LDS
  if (POW2) __builtin_amdgcn_s_wait_asynccnt(0);
#endif
  __syncthreads();

  int buf = 0;
  const int mrow  = lane & 15;
  const int kgrpA = (lane >> 4) << 3;   // 0 or 8
  const int kgrpB = (lane >> 4) << 4;   // 0 or 16

  for (int k0 = 0; k0 < K; k0 += TK) {
    if (k0 + TK < K) load_stage(k0 + TK, buf ^ 1);

    FragA a[2], b[2];
    #pragma unroll
    for (int mi = 0; mi < 2; ++mi) {
      const int row = 32 * wm + 16 * mi + mrow;
      #pragma unroll
      for (int e = 0; e < 16; ++e) {
        const int k = (e & 7) + ((e >> 3) << 4) + kgrpA;  // {0..7,16..23}(+8)
        a[mi].h[e] = As[buf][row][k];
      }
    }
    #pragma unroll
    for (int ni = 0; ni < 2; ++ni) {
      const int col = 32 * wn + 16 * ni + mrow;
      #pragma unroll
      for (int e = 0; e < 16; ++e)
        b[ni].h[e] = Bs[buf][(e & 7) + ((e >> 3) << 4) + kgrpA][col];  // K {0..7,16..23}(+8)
    }

    #pragma unroll
    for (int mi = 0; mi < 2; ++mi)
      #pragma unroll
      for (int ni = 0; ni < 2; ++ni)
        acc[mi][ni].v = __builtin_amdgcn_wmma_f32_16x16x32_f16(
            false, a[mi].v, false, b[ni].v, (short)0, acc[mi][ni].v, false, false);

#if HAVE_ASYNC_LDS
    if (POW2) __builtin_amdgcn_s_wait_asynccnt(0);
#endif
    __syncthreads();
    buf ^= 1;
  }

  // ---- epilogue: BN scale/bias (+ conv bias folded), residual, ReLU ----
  #pragma unroll
  for (int mi = 0; mi < 2; ++mi) {
    #pragma unroll
    for (int ni = 0; ni < 2; ++ni) {
      const int ncol = n0 + 32 * wn + 16 * ni + (lane & 15);
      const float sc = scale[ncol];
      const float bi = bias[ncol];
      #pragma unroll
      for (int r = 0; r < 8; ++r) {
        const int mg = m0 + 32 * wm + 16 * mi + r + ((lane >> 4) << 3);
        float v = acc[mi][ni].f[r] * sc + bi;
        const size_t off = (size_t)mg * Cout + ncol;
        if (resid) v += (float)resid[off];
        if (relu)  v = fmaxf(v, 0.f);
        out[off] = (_Float16)v;
      }
    }
  }
}

// ------------------------------- host driver -------------------------------

extern "C" void kernel_launch(void* const* d_in, const int* in_sizes, int n_in,
                              void* d_out, int out_size, void* d_ws, size_t ws_size,
                              hipStream_t stream) {
  (void)in_sizes; (void)n_in; (void)out_size; (void)ws_size;
  const int NB = 256;

  // deterministic bump allocator on workspace (~122 MB total)
  char* ws = (char*)d_ws;
  auto alloc = [&](size_t bytes) -> char* {
    char* r = ws;
    ws += (bytes + 255) & ~(size_t)255;
    return r;
  };

  _Float16* xh   = (_Float16*)alloc((size_t)NB * 32 * 32 * 3 * 2);
  _Float16* bufA = (_Float16*)alloc((size_t)NB * 32 * 32 * 64 * 2);
  _Float16* bufB = (_Float16*)alloc((size_t)NB * 32 * 32 * 64 * 2);
  _Float16* bufC = (_Float16*)alloc((size_t)NB * 32 * 32 * 64 * 2);
  float*    amax = (float*)alloc(256);

  // param cursor — setup_inputs() dict insertion order, depth-first:
  // x; stem{w,b,bn{gamma,beta,mean,var}};
  // layers[li][bi]{w1,b1,bn1{g,b,m,v}, w2,b2,bn2{g,b,m,v} [,wd,bd,bnd{g,b,m,v}]};
  // fc{w,b}
  int pi = 0;
  auto nxt = [&]() { return (const float*)d_in[pi++]; };

  auto run_conv = [&](const _Float16* in, _Float16* out, const _Float16* resid,
                      const float* w, const float* bconv,
                      const float* g, const float* be, const float* mu, const float* var,
                      int H, int Cin, int Cout, int KHW, int stride, int pad,
                      int relu, int quant) {
    const int W  = H;
    const int Ho = (H + 2 * pad - KHW) / stride + 1;
    const int Wo = Ho;
    const int K  = Cin * KHW * KHW;
    const int nW = Cout * K;
    _Float16* wq = (_Float16*)alloc((size_t)nW * 2);
    float* sc = (float*)alloc((size_t)Cout * 4);
    float* bi = (float*)alloc((size_t)Cout * 4);

    if (quant)
      absmax_scale_kernel<<<1, 1024, 0, stream>>>(w, nW, amax);
    quant_pack_kernel<<<(nW + 255) / 256, 256, 0, stream>>>(w, amax, wq, Cout, Cin, KHW, KHW, quant);
    bn_fold_kernel<<<(Cout + 255) / 256, 256, 0, stream>>>(bconv, g, be, mu, var, sc, bi, Cout);

    dim3 grid((unsigned)((NB * Ho * Wo) / TM), (unsigned)(Cout / TN));
    const int cinShift = ((Cin & (Cin - 1)) == 0) ? __builtin_ctz(Cin) : -1;
    const int hwShift  = __builtin_ctz(Ho * Wo);   // Ho*Wo always pow2 here
    const int woShift  = __builtin_ctz(Wo);
    if (KHW == 3) {
      if (cinShift >= 0)
        conv_wmma_kernel<3, true><<<grid, 256, 0, stream>>>(
            in, wq, sc, bi, resid, out, H, W, Cin, Cout, stride, pad,
            relu, cinShift, hwShift, woShift);
      else
        conv_wmma_kernel<3, false><<<grid, 256, 0, stream>>>(
            in, wq, sc, bi, resid, out, H, W, Cin, Cout, stride, pad,
            relu, cinShift, hwShift, woShift);
    } else {
      conv_wmma_kernel<1, true><<<grid, 256, 0, stream>>>(
          in, wq, sc, bi, resid, out, H, W, Cin, Cout, stride, pad,
          relu, cinShift, hwShift, woShift);
    }
  };

  // ---- input -> NHWC f16 ----
  const float* x = nxt();
  {
    int total = NB * 3 * 32 * 32;
    nchw_to_nhwc_h_kernel<<<(total + 255) / 256, 256, 0, stream>>>(x, xh, NB, 3, 32, 32);
  }

  // ---- stem: plain conv3x3 s1 p1 + bn + relu, then maxpool 3x3 s2 p1 ----
  {
    const float* sw = nxt(); const float* sb = nxt();
    const float* sg = nxt(); const float* sbe = nxt();
    const float* smu = nxt(); const float* svar = nxt();
    run_conv(xh, bufA, nullptr, sw, sb, sg, sbe, smu, svar,
             /*H*/32, /*Cin*/3, /*Cout*/64, /*KHW*/3, /*stride*/1, /*pad*/1,
             /*relu*/1, /*quant*/0);
    int total = NB * 16 * 16 * 64;
    maxpool3x3s2_kernel<<<(total + 255) / 256, 256, 0, stream>>>(bufA, bufB, NB, 32, 32, 64, 16, 16);
  }

  // ---- residual stages ----
  _Float16 *cur = bufB, *s0 = bufA, *s1 = bufC;
  const int chans[4] = {64, 128, 256, 512};
  int cin = 64, Hcur = 16;
  for (int li = 0; li < 4; ++li) {
    const int cout = chans[li];
    for (int bi2 = 0; bi2 < 2; ++bi2) {
      const int stride = (bi2 == 0 && li > 0) ? 2 : 1;
      const bool down  = (bi2 == 0 && li > 0);
      const float *w1 = nxt(), *b1 = nxt(), *g1 = nxt(), *be1 = nxt(), *mu1 = nxt(), *v1 = nxt();
      const float *w2 = nxt(), *b2 = nxt(), *g2 = nxt(), *be2 = nxt(), *mu2 = nxt(), *v2 = nxt();
      const float *wd = nullptr, *bd = nullptr, *gd = nullptr, *bed = nullptr, *mud = nullptr, *vd = nullptr;
      if (down) { wd = nxt(); bd = nxt(); gd = nxt(); bed = nxt(); mud = nxt(); vd = nxt(); }
      const int Hout = Hcur / stride;
      if (down) {
        // identity path: 1x1 conv stride s + bn (no relu)
        run_conv(cur, s0, nullptr, wd, bd, gd, bed, mud, vd, Hcur, cin, cout, 1, stride, 0, 0, 1);
        // main path
        run_conv(cur, s1, nullptr, w1, b1, g1, be1, mu1, v1, Hcur, cin, cout, 3, stride, 1, 1, 1);
        run_conv(s1, cur, s0,      w2, b2, g2, be2, mu2, v2, Hout, cout, cout, 3, 1, 1, 1, 1);
        // block output in `cur`
      } else {
        run_conv(cur, s0, nullptr, w1, b1, g1, be1, mu1, v1, Hcur, cin, cout, 3, 1, 1, 1, 1);
        run_conv(s0, s1, cur,      w2, b2, g2, be2, mu2, v2, Hcur, cout, cout, 3, 1, 1, 1, 1);
        _Float16* old = cur; cur = s1; s1 = old;
      }
      cin = cout; Hcur = Hout;
    }
  }

  // ---- avgpool(2x2) + fc(512 -> 10) ----
  {
    const float* fw = nxt();
    const float* fb = nxt();
    avgpool_fc_kernel<<<NB, 128, 0, stream>>>(cur, fw, fb, (float*)d_out, /*HW*/4, /*C*/512, /*ncls*/10);
  }
}
